// MultiHeadAttentionLayer_27212912788246
// MI455X (gfx1250) — compile-verified, matrix operand order checked by reading, not run on hardware
//
#include <hip/hip_runtime.h>
#include <hip/hip_bf16.h>
#include <math.h>

#define B_      8
#define N_      1024
#define IN_DIM_ 512
#define HID_    128
#define NH_     8
#define OUT_DIM_ 512
#define NEG_BIG_F 9000000000000000.0f

typedef __attribute__((ext_vector_type(16))) _Float16 v16h;
typedef __attribute__((ext_vector_type(8)))  _Float16 v8h;
typedef __attribute__((ext_vector_type(8)))  float    v8f;

// ---------------------------------------------------------------------------
// WMMA fragment loaders (CDNA5 16-bit A/B layout: lane&15 = row(M)/col(N),
// halves 0..7 = K = 8*(lane>>4)+0..7, halves 8..15 = K = 16+8*(lane>>4)+0..7)
// Source matrix stored row-major with K contiguous (ld = K-stride in halves).
// ---------------------------------------------------------------------------
__device__ __forceinline__ v16h load_frag_f16(const _Float16* __restrict__ base,
                                              int ld, int row0, int kk) {
  int lane = threadIdx.x & 31;
  const _Float16* p = base + (size_t)(row0 + (lane & 15)) * ld + kk + ((lane >> 4) << 3);
  v8h lo = *(const v8h*)(p);
  v8h hi = *(const v8h*)(p + 16);
  return __builtin_shufflevector(lo, hi, 0,1,2,3,4,5,6,7,8,9,10,11,12,13,14,15);
}

// Same fragment but sourced from fp32 memory (attention probs) with in-register
// cvt to f16.
__device__ __forceinline__ v16h load_frag_f32cvt(const float* __restrict__ base,
                                                 int ld, int row0, int kk) {
  int lane = threadIdx.x & 31;
  const float* p = base + (size_t)(row0 + (lane & 15)) * ld + kk + ((lane >> 4) << 3);
  float4 a0 = *(const float4*)(p);
  float4 a1 = *(const float4*)(p + 4);
  float4 b0 = *(const float4*)(p + 16);
  float4 b1 = *(const float4*)(p + 20);
  v16h f;
  f[0]  = (_Float16)a0.x; f[1]  = (_Float16)a0.y; f[2]  = (_Float16)a0.z; f[3]  = (_Float16)a0.w;
  f[4]  = (_Float16)a1.x; f[5]  = (_Float16)a1.y; f[6]  = (_Float16)a1.z; f[7]  = (_Float16)a1.w;
  f[8]  = (_Float16)b0.x; f[9]  = (_Float16)b0.y; f[10] = (_Float16)b0.z; f[11] = (_Float16)b0.w;
  f[12] = (_Float16)b1.x; f[13] = (_Float16)b1.y; f[14] = (_Float16)b1.z; f[15] = (_Float16)b1.w;
  return f;
}

// ---------------------------------------------------------------------------
// Elementwise converters
// ---------------------------------------------------------------------------
__global__ void cvt_f32_to_f16_kernel(const float* __restrict__ src,
                                      _Float16* __restrict__ dst, int n) {
  int i = blockIdx.x * blockDim.x + threadIdx.x;
  if (i < n) dst[i] = (_Float16)src[i];
}

// Wt[n*K + k] = (f16) W[k*Ncol + n]   (store weights K-contiguous per output col)
__global__ void transpose_cvt_kernel(const float* __restrict__ W,
                                     _Float16* __restrict__ Wt, int K, int Ncol) {
  int i = blockIdx.x * blockDim.x + threadIdx.x;
  if (i < K * Ncol) {
    int n = i / K;
    int k = i - n * K;
    Wt[i] = (_Float16)W[(size_t)k * Ncol + n];
  }
}

// ---------------------------------------------------------------------------
// WMMA GEMM:  y = relu(A @ B + bias),  A: MxK f16 row-major, Bt: Ncol x K f16.
// One 16x16 output tile per wave; all operands L2-resident at these sizes.
// mode 0: f16 store as (b, h, n, d)          [q, k]
// mode 1: f16 store as (b, h, d, n)          [v transposed for the AV GEMM]
// mode 2: f32 store row-major to d_out       [final output projection]
// ---------------------------------------------------------------------------
__global__ void gemm_relu_kernel(const _Float16* __restrict__ A,
                                 const _Float16* __restrict__ Bt,
                                 const float* __restrict__ bias,
                                 _Float16* __restrict__ outh,
                                 float* __restrict__ outf,
                                 int M, int K, int Ncol, int mode) {
  int wave   = blockIdx.x * (blockDim.x >> 5) + (threadIdx.x >> 5);
  int tilesN = Ncol >> 4;
  int tm = wave / tilesN;
  int tn = wave - tm * tilesN;
  if (tm * 16 >= M) return;
  int lane = threadIdx.x & 31;

  v8f acc = {};
  for (int kk = 0; kk < K; kk += 32) {
    v16h a = load_frag_f16(A,  K, tm * 16, kk);
    v16h b = load_frag_f16(Bt, K, tn * 16, kk);
    acc = __builtin_amdgcn_wmma_f32_16x16x32_f16(false, a, false, b,
                                                 (short)0, acc, false, false);
  }

  int col   = tn * 16 + (lane & 15);
  float bv  = bias[col];
  int rbase = tm * 16 + ((lane >> 4) << 3);
#pragma unroll
  for (int r = 0; r < 8; ++r) {
    float y = acc[r] + bv;
    y = y > 0.0f ? y : 0.0f;
    int row = rbase + r;
    if (mode == 2) {
      outf[(size_t)row * Ncol + col] = y;
    } else {
      int bb = row >> 10;            // row / N_
      int n  = row & (N_ - 1);
      int hh = col >> 7;             // col / HID_
      int d  = col & (HID_ - 1);
      _Float16 hv = (_Float16)y;
      size_t base = (size_t)(bb * NH_ + hh);
      if (mode == 0) outh[(base * N_ + n) * HID_ + d] = hv;   // (b,h,n,d)
      else           outh[(base * HID_ + d) * N_ + n] = hv;   // (b,h,d,n)
    }
  }
}

// ---------------------------------------------------------------------------
// Fused QK^T + mask + softmax. One wave owns 16 query rows for one (b,h).
// Pass 1 accumulates row sums of exp(masked logits) (logits are O(1) after
// 1/sqrt(128) scaling, masked entries are -9e15 -> exp == 0, so no running
// max is needed); pass 2 recomputes tiles (L2-fed, compute is free) and
// writes normalized probs in attn_cat layout [h*B+b, n, m] (fp32, to d_out).
// ---------------------------------------------------------------------------
__global__ void attn_softmax_kernel(const _Float16* __restrict__ qh,
                                    const _Float16* __restrict__ kh,
                                    const float* __restrict__ mask,
                                    float* __restrict__ attn) {
  int bh   = blockIdx.x;                 // b*NH + h
  int b    = bh >> 3;                    // / NH_
  int h    = bh & 7;
  int wave = threadIdx.x >> 5;           // 4 waves/block
  int row0 = blockIdx.y * 64 + wave * 16;
  int lane = threadIdx.x & 31;

  const _Float16* qbase = qh + (size_t)bh * N_ * HID_;
  const _Float16* kbase = kh + (size_t)bh * N_ * HID_;
  const float*    mbase = mask + (size_t)b * N_ * N_;
  float*          abase = attn + (size_t)(h * B_ + b) * N_ * N_;

  v16h qf[4];
#pragma unroll
  for (int t = 0; t < 4; ++t) qf[t] = load_frag_f16(qbase, HID_, row0, t * 32);

  const float scale = 0.08838834764831845f;   // 1/sqrt(HID)
  int rbase = row0 + ((lane >> 4) << 3);
  int csub  = lane & 15;

  float rsum[8];
#pragma unroll
  for (int r = 0; r < 8; ++r) rsum[r] = 0.0f;

  // pass 1: row sums of exp(masked logits)
  for (int mt = 0; mt < (N_ >> 4); ++mt) {
    v8f acc = {};
#pragma unroll
    for (int t = 0; t < 4; ++t) {
      v16h kf = load_frag_f16(kbase, HID_, mt * 16, t * 32);
      acc = __builtin_amdgcn_wmma_f32_16x16x32_f16(false, qf[t], false, kf,
                                                   (short)0, acc, false, false);
    }
    int col = mt * 16 + csub;
#pragma unroll
    for (int r = 0; r < 8; ++r) {
      float m = mbase[(size_t)(rbase + r) * N_ + col];
      float s = acc[r] * scale * m - NEG_BIG_F * (1.0f - m);
      rsum[r] += __expf(s);
    }
  }
  // butterfly reduce across the 16 lanes sharing each row (each half-wave
  // owns 8 rows; xor masks 1,2,4,8 stay within the half)
#pragma unroll
  for (int r = 0; r < 8; ++r) {
    float v = rsum[r];
    v += __shfl_xor(v, 1, 32);
    v += __shfl_xor(v, 2, 32);
    v += __shfl_xor(v, 4, 32);
    v += __shfl_xor(v, 8, 32);
    rsum[r] = 1.0f / v;
  }
  // pass 2: recompute + write normalized attention
  for (int mt = 0; mt < (N_ >> 4); ++mt) {
    v8f acc = {};
#pragma unroll
    for (int t = 0; t < 4; ++t) {
      v16h kf = load_frag_f16(kbase, HID_, mt * 16, t * 32);
      acc = __builtin_amdgcn_wmma_f32_16x16x32_f16(false, qf[t], false, kf,
                                                   (short)0, acc, false, false);
    }
    int col = mt * 16 + csub;
#pragma unroll
    for (int r = 0; r < 8; ++r) {
      float m = mbase[(size_t)(rbase + r) * N_ + col];
      float s = acc[r] * scale * m - NEG_BIG_F * (1.0f - m);
      abase[(size_t)(rbase + r) * N_ + col] = __expf(s) * rsum[r];
    }
  }
}

// ---------------------------------------------------------------------------
// out_h[b,n,h*HID+d] = sum_m attn[h*B+b,n,m] * v[b,m,h,d]
// A = attn (fp32 -> f16 in-register), B = V^T (d x m, K-contiguous). One
// 16(n) x 16(d) tile per wave, K loop over m.
// ---------------------------------------------------------------------------
__global__ void attn_v_kernel(const float* __restrict__ attn,
                              const _Float16* __restrict__ vt,
                              _Float16* __restrict__ oh) {
  int wave = blockIdx.x * (blockDim.x >> 5) + (threadIdx.x >> 5);
  int bh = wave >> 9;            // 512 tiles per (b,h): 64 n-tiles x 8 d-tiles
  int t  = wave & 511;
  int nt = t >> 3;
  int dt = t & 7;
  int b  = bh >> 3;
  int h  = bh & 7;
  int lane = threadIdx.x & 31;

  const float*    abase = attn + (size_t)(h * B_ + b) * N_ * N_;
  const _Float16* vbase = vt + (size_t)bh * HID_ * N_;

  v8f acc = {};
  for (int kk = 0; kk < N_; kk += 32) {
    v16h af = load_frag_f32cvt(abase, N_, nt * 16, kk);
    v16h bf = load_frag_f16(vbase, N_, dt * 16, kk);
    acc = __builtin_amdgcn_wmma_f32_16x16x32_f16(false, af, false, bf,
                                                 (short)0, acc, false, false);
  }

  int col   = h * HID_ + dt * 16 + (lane & 15);
  int rbase = b * N_ + nt * 16 + ((lane >> 4) << 3);
#pragma unroll
  for (int r = 0; r < 8; ++r)
    oh[(size_t)(rbase + r) * (NH_ * HID_) + col] = (_Float16)acc[r];
}

// ---------------------------------------------------------------------------
extern "C" void kernel_launch(void* const* d_in, const int* in_sizes, int n_in,
                              void* d_out, int out_size, void* d_ws, size_t ws_size,
                              hipStream_t stream) {
  (void)in_sizes; (void)n_in; (void)out_size; (void)ws_size;

  const float* x    = (const float*)d_in[0];
  const float* mask = (const float*)d_in[1];
  const float* Wq   = (const float*)d_in[2];
  const float* bq   = (const float*)d_in[3];
  const float* Wk   = (const float*)d_in[4];
  const float* bk   = (const float*)d_in[5];
  const float* Wv   = (const float*)d_in[6];
  const float* bv   = (const float*)d_in[7];
  const float* Wo   = (const float*)d_in[8];
  const float* bo   = (const float*)d_in[9];

  float* outp = (float*)d_out;                         // (B, N, OUT_DIM)
  float* attn = outp + (size_t)B_ * N_ * OUT_DIM_;     // (NH*B, N, N)

  // workspace carve-up (f16 elements); total ~80 MB
  _Float16* ws = (_Float16*)d_ws;
  size_t o = 0;
  _Float16* xh  = ws + o; o += (size_t)B_ * N_ * IN_DIM_;        // x  -> f16
  _Float16* wqt = ws + o; o += (size_t)(NH_ * HID_) * IN_DIM_;   // Wq^T f16
  _Float16* wkt = ws + o; o += (size_t)(NH_ * HID_) * IN_DIM_;
  _Float16* wvt = ws + o; o += (size_t)(NH_ * HID_) * IN_DIM_;
  _Float16* wot = ws + o; o += (size_t)OUT_DIM_ * (NH_ * HID_);  // Wo^T f16
  _Float16* qh  = ws + o; o += (size_t)B_ * NH_ * N_ * HID_;     // (b,h,n,d)
  _Float16* kh  = ws + o; o += (size_t)B_ * NH_ * N_ * HID_;     // (b,h,m,d)
  _Float16* vt  = ws + o; o += (size_t)B_ * NH_ * HID_ * N_;     // (b,h,d,m)
  _Float16* oh  = ws + o;                                         // (b*n, h*d)

  const int nX = B_ * N_ * IN_DIM_;          // 4,194,304
  const int nW = IN_DIM_ * NH_ * HID_;       //   524,288

  cvt_f32_to_f16_kernel<<<(nX + 255) / 256, 256, 0, stream>>>(x, xh, nX);
  transpose_cvt_kernel<<<(nW + 255) / 256, 256, 0, stream>>>(Wq, wqt, IN_DIM_, NH_ * HID_);
  transpose_cvt_kernel<<<(nW + 255) / 256, 256, 0, stream>>>(Wk, wkt, IN_DIM_, NH_ * HID_);
  transpose_cvt_kernel<<<(nW + 255) / 256, 256, 0, stream>>>(Wv, wvt, IN_DIM_, NH_ * HID_);
  transpose_cvt_kernel<<<(nW + 255) / 256, 256, 0, stream>>>(Wo, wot, NH_ * HID_, OUT_DIM_);

  // QKV projections: M=8192, K=512, Ncol=1024 -> 32768 tiles / 8 waves = 4096 blocks
  gemm_relu_kernel<<<4096, 256, 0, stream>>>(xh, wqt, bq, qh, nullptr, B_ * N_, IN_DIM_, NH_ * HID_, 0);
  gemm_relu_kernel<<<4096, 256, 0, stream>>>(xh, wkt, bk, kh, nullptr, B_ * N_, IN_DIM_, NH_ * HID_, 0);
  gemm_relu_kernel<<<4096, 256, 0, stream>>>(xh, wvt, bv, vt, nullptr, B_ * N_, IN_DIM_, NH_ * HID_, 1);

  // Fused logits + mask + softmax: 64 (b,h) x 16 row-blocks, 4 waves/block
  attn_softmax_kernel<<<dim3(B_ * NH_, N_ / 64), 128, 0, stream>>>(qh, kh, mask, attn);

  // attn @ V : 32768 tiles / 8 waves = 4096 blocks
  attn_v_kernel<<<4096, 256, 0, stream>>>(attn, vt, oh);

  // Output projection: M=8192, K=1024, Ncol=512 -> 16384 tiles -> 2048 blocks
  gemm_relu_kernel<<<2048, 256, 0, stream>>>(oh, wot, bo, nullptr, outp, B_ * N_, NH_ * HID_, OUT_DIM_, 2);
}
